// RwkvSelfAttention_25305947308163
// MI455X (gfx1250) — matrix operand order repaired and checked by reading, not compiled, not run on hardware
//
#include <hip/hip_runtime.h>

typedef __bf16 bf16_t;
typedef __attribute__((ext_vector_type(16))) __bf16 v16bf;
typedef __attribute__((ext_vector_type(8)))  __bf16 v8bf;
typedef __attribute__((ext_vector_type(8)))  float  v8f;
typedef __attribute__((ext_vector_type(4)))  unsigned int u32x4;
typedef __attribute__((ext_vector_type(8)))  int          i32x8;
typedef __attribute__((ext_vector_type(4)))  int          i32x4;

#define H_DIM 2048
#define T_DIM 8192

// GEMM block tile: 64 M x 128 N, K-step 32 (one bf16 WMMA K-slice)
#define BLK_M 64
#define BLK_N 128
#define KSTEP 32
#define PITCH 40                   // LDS row pitch in bf16 elems: 64B data + 16B pad
#define A_ELEMS (BLK_M * PITCH)
#define B_ELEMS (BLK_N * PITCH)
#define BUF_ELEMS (A_ELEMS + B_ELEMS)

#if __has_builtin(__builtin_amdgcn_tensor_load_to_lds) && \
    __has_builtin(__builtin_amdgcn_s_wait_tensorcnt)
#define HAVE_TDM 1
#else
#define HAVE_TDM 0
#endif

union ABFrag { v16bf v; v8bf h[2]; };

__device__ __forceinline__ int rfl(unsigned x) {
  return __builtin_amdgcn_readfirstlane((int)x);
}

#if HAVE_TDM
// Issue one TDM 2D tile load: rows x KSTEP bf16 elements, global row stride
// H_DIM elems, into LDS at lds_byte_off with 16B padding after each 64B row.
__device__ __forceinline__ void tdm_load_tile(const bf16_t* gsrc,
                                              unsigned lds_byte_off, int rows) {
  unsigned long long ga = (unsigned long long)(uintptr_t)gsrc;
  // --- D# group 0 (128b): count=1 | lds_addr | global_addr | type=2
  unsigned g00 = 1u;                                   // count=1, user desc
  unsigned g01 = lds_byte_off;                         // lds_addr
  unsigned g02 = (unsigned)ga;                         // global_addr[31:0]
  unsigned g03 = (unsigned)(ga >> 32) | (2u << 30);    // ga[56:32] | type=2
  // --- D# group 1 (256b)
  unsigned w0 = (1u << 16)      // data_size = 2 bytes
              | (1u << 20)      // pad_enable
              | (3u << 22)      // pad_interval: 16 dwords (one 64B row)
              | (3u << 25);     // pad_amount:  4 dwords (16B)
  unsigned td0 = H_DIM;                 // tensor_dim0 (elems)
  unsigned td1 = (unsigned)rows;        // tensor_dim1 (rows)
  unsigned w1 = (td0 & 0xFFFFu) << 16;                 // [47:32]=0 (no atomic barrier)
  unsigned w2 = (td0 >> 16) | ((td1 & 0xFFFFu) << 16);
  unsigned w3 = (td1 >> 16) | ((unsigned)KSTEP << 16); // tile_dim0 = 32 elems
  unsigned w4 = (unsigned)rows;                        // tile_dim1; tile_dim2=0
  unsigned w5 = H_DIM;                                 // tensor_dim0_stride[31:0]
  unsigned w6 = 0u;
  unsigned w7 = 0u;

  u32x4 g0 = { (unsigned)rfl(g00), (unsigned)rfl(g01),
               (unsigned)rfl(g02), (unsigned)rfl(g03) };
  i32x8 g1 = { rfl(w0), rfl(w1), rfl(w2), rfl(w3),
               rfl(w4), rfl(w5), rfl(w6), rfl(w7) };
  i32x4 z4 = { 0, 0, 0, 0 };
  i32x8 z8 = { 0, 0, 0, 0, 0, 0, 0, 0 };
  __builtin_amdgcn_tensor_load_to_lds(g0, g1, z4, z4, z8, 0);
}
#endif

// ---------------------------------------------------------------------------
// 1) Token-shift mix: kx/vx/rx = x*mix + x_prev*(1-mix), stored bf16 for WMMA
// ---------------------------------------------------------------------------
__global__ __launch_bounds__(256) void mix_kernel(
    const float* __restrict__ hidden, const float* __restrict__ sx,
    const float* __restrict__ tmk, const float* __restrict__ tmv,
    const float* __restrict__ tmr,
    bf16_t* __restrict__ kx, bf16_t* __restrict__ vx, bf16_t* __restrict__ rx) {
  int idx = blockIdx.x * blockDim.x + threadIdx.x;
  if (idx >= T_DIM * H_DIM) return;
  int h = idx & (H_DIM - 1);
  float x = hidden[idx];
  float c = (idx < H_DIM) ? sx[h] : hidden[idx - H_DIM];
  float mk = tmk[h], mv = tmv[h], mr = tmr[h];
  kx[idx] = (bf16_t)(x * mk + c * (1.0f - mk));
  vx[idx] = (bf16_t)(x * mv + c * (1.0f - mv));
  rx[idx] = (bf16_t)(x * mr + c * (1.0f - mr));
}

// ---------------------------------------------------------------------------
// 2) Weight transpose f32[K,N] -> bf16[N,K] via LDS tile (coalesced both ways)
// ---------------------------------------------------------------------------
__global__ __launch_bounds__(256) void transpose_kernel(
    const float* __restrict__ Wk, const float* __restrict__ Wv,
    const float* __restrict__ Wr, const float* __restrict__ Wo,
    bf16_t* __restrict__ WkT, bf16_t* __restrict__ WvT,
    bf16_t* __restrict__ WrT, bf16_t* __restrict__ WoT) {
  const float* W; bf16_t* WT;
  switch (blockIdx.z) {
    case 0:  W = Wk; WT = WkT; break;
    case 1:  W = Wv; WT = WvT; break;
    case 2:  W = Wr; WT = WrT; break;
    default: W = Wo; WT = WoT; break;
  }
  __shared__ float tile[32][33];
  int tx = threadIdx.x & 31;
  int ty = threadIdx.x >> 5;          // 0..7
  int n0 = blockIdx.x * 32;
  int k0 = blockIdx.y * 32;
#pragma unroll
  for (int i = 0; i < 32; i += 8)
    tile[ty + i][tx] = W[(size_t)(k0 + ty + i) * H_DIM + n0 + tx];
  __syncthreads();
#pragma unroll
  for (int i = 0; i < 32; i += 8)
    WT[(size_t)(n0 + ty + i) * H_DIM + (k0 + tx)] = (bf16_t)tile[tx][ty + i];
}

// ---------------------------------------------------------------------------
// 3) GEMM: C[M,N] = A[M,K] * BT[N,K]^T   (A,BT bf16; C f32)
//    Block = 8 waves, block tile 64M x 128N, K-step 32.
//    TDM double-buffers A/B tiles in LDS; waves consume via ds_load + WMMA.
//    Wave tile: 16M x 64N (4 accumulators); wm = wave&3, wn = wave>>2.
//    All fragment ds_loads are issued before the WMMA burst so only one
//    s_wait_dscnt is needed per K-step.
//    mode 0: plain store   mode 1: sigmoid   mode 2: add residual
// ---------------------------------------------------------------------------
__global__ __launch_bounds__(256) void gemm_kernel(
    const bf16_t* __restrict__ A, const bf16_t* __restrict__ BT,
    float* __restrict__ C, const float* __restrict__ resid, int mode) {
  __shared__ __align__(16) bf16_t smem[2 * BUF_ELEMS];

  int lane = threadIdx.x & 31;
  int wave = threadIdx.x >> 5;
  int wm = wave & 3;                  // 16-row band within 64
  int wn = wave >> 2;                 // 64-col band within 128
  int half = lane >> 4;               // 0 or 1
  int l16  = lane & 15;

  int mblock = blockIdx.y * BLK_M;
  int nblock = blockIdx.x * BLK_N;
  const bf16_t* gA = A  + (size_t)mblock * H_DIM;
  const bf16_t* gB = BT + (size_t)nblock * H_DIM;

  v8f acc[4] = {v8f{}, v8f{}, v8f{}, v8f{}};

  int arow = wm * 16 + l16;

#if HAVE_TDM
  unsigned lds_base = (unsigned)(uintptr_t)(void*)&smem[0];
  if (wave == 0) {
    tdm_load_tile(gA, lds_base, BLK_M);
    tdm_load_tile(gB, lds_base + A_ELEMS * 2, BLK_N);
  }
  int buf = 0;
  for (int k0 = 0; k0 < H_DIM; k0 += KSTEP) {
    if (wave == 0) __builtin_amdgcn_s_wait_tensorcnt(0);
    __syncthreads();                  // current buf ready; prev buf free
    if (wave == 0 && (k0 + KSTEP) < H_DIM) {
      unsigned off = (unsigned)((buf ^ 1) * BUF_ELEMS * 2);
      tdm_load_tile(gA + k0 + KSTEP, lds_base + off, BLK_M);
      tdm_load_tile(gB + k0 + KSTEP, lds_base + off + A_ELEMS * 2, BLK_N);
    }
    const bf16_t* As = &smem[buf * BUF_ELEMS];
    const bf16_t* Bs = As + A_ELEMS;

    // Issue ALL fragment loads first, then the WMMA burst.
    ABFrag a;
    a.h[0] = *(const v8bf*)&As[arow * PITCH + 16 * half];
    a.h[1] = *(const v8bf*)&As[arow * PITCH + 16 * half + 8];
    ABFrag b[4];
#pragma unroll
    for (int j = 0; j < 4; ++j) {
      int brow = wn * 64 + 16 * j + l16;
      b[j].h[0] = *(const v8bf*)&Bs[brow * PITCH + 16 * half];
      b[j].h[1] = *(const v8bf*)&Bs[brow * PITCH + 16 * half + 8];
    }
#pragma unroll
    for (int j = 0; j < 4; ++j)
      acc[j] = __builtin_amdgcn_wmma_f32_16x16x32_bf16(
          false, a.v, false, b[j].v, (short)0, acc[j], false, false);
    buf ^= 1;
  }
#else
  // Fallback: cooperative copy staging (single buffer)
  for (int k0 = 0; k0 < H_DIM; k0 += KSTEP) {
    __syncthreads();
    for (int i = threadIdx.x; i < BLK_M * KSTEP; i += 256) {
      int r = i >> 5, c = i & 31;
      smem[r * PITCH + c] = gA[(size_t)r * H_DIM + k0 + c];
    }
    for (int i = threadIdx.x; i < BLK_N * KSTEP; i += 256) {
      int r = i >> 5, c = i & 31;
      smem[A_ELEMS + r * PITCH + c] = gB[(size_t)r * H_DIM + k0 + c];
    }
    __syncthreads();
    const bf16_t* As = &smem[0];
    const bf16_t* Bs = &smem[A_ELEMS];
    ABFrag a;
    a.h[0] = *(const v8bf*)&As[arow * PITCH + 16 * half];
    a.h[1] = *(const v8bf*)&As[arow * PITCH + 16 * half + 8];
    ABFrag b[4];
#pragma unroll
    for (int j = 0; j < 4; ++j) {
      int brow = wn * 64 + 16 * j + l16;
      b[j].h[0] = *(const v8bf*)&Bs[brow * PITCH + 16 * half];
      b[j].h[1] = *(const v8bf*)&Bs[brow * PITCH + 16 * half + 8];
    }
#pragma unroll
    for (int j = 0; j < 4; ++j)
      acc[j] = __builtin_amdgcn_wmma_f32_16x16x32_bf16(
          false, a.v, false, b[j].v, (short)0, acc[j], false, false);
  }
#endif

#pragma unroll
  for (int j = 0; j < 4; ++j) {
    int n = nblock + wn * 64 + 16 * j + l16;
#pragma unroll
    for (int v = 0; v < 8; ++v) {
      int m = mblock + wm * 16 + v + 8 * half;  // C layout: VGPR v -> M
      float val = acc[j][v];
      if (mode == 1)      val = 1.0f / (1.0f + __expf(-val));
      else if (mode == 2) val += resid[(size_t)m * H_DIM + n];
      C[(size_t)m * H_DIM + n] = val;
    }
  }
}

// ---------------------------------------------------------------------------
// 4) WKV scan: serial in T, parallel across H channels (2048 lanes).
// ---------------------------------------------------------------------------
__global__ __launch_bounds__(256) void scan_kernel(
    const float* __restrict__ kf, const float* __restrict__ vf,
    const float* __restrict__ rf,
    const float* __restrict__ aa0, const float* __restrict__ bb0,
    const float* __restrict__ pp0,
    const float* __restrict__ time_decay, const float* __restrict__ time_first,
    const float* __restrict__ hidden,
    bf16_t* __restrict__ rwkv, float* __restrict__ out_states) {
  int c = blockIdx.x * blockDim.x + threadIdx.x;
  if (c >= H_DIM) return;
  float a = aa0[c], b = bb0[c], p = pp0[c];
  float tf = time_first[c];
  float wd = -__expf(time_decay[c]);
  for (int t = 0; t < T_DIM; ++t) {
    size_t idx = (size_t)t * H_DIM + c;
    float kk = kf[idx], vv = vf[idx];
    float ww = tf + kk;
    float q  = fmaxf(p, ww);
    float e1 = __expf(p - q);
    float e2 = __expf(ww - q);
    float wkv = (e1 * a + e2 * vv) / (e1 * b + e2);
    rwkv[idx] = (bf16_t)(rf[idx] * wkv);
    float ww2 = wd + p;
    float q2  = fmaxf(ww2, kk);
    float e1b = __expf(ww2 - q2);
    float e2b = __expf(kk - q2);
    a = e1b * a + e2b * vv;
    b = e1b * b + e2b;
    p = q2;
  }
  out_states[c]             = hidden[(size_t)(T_DIM - 1) * H_DIM + c];
  out_states[H_DIM + c]     = a;
  out_states[2 * H_DIM + c] = b;
  out_states[3 * H_DIM + c] = p;
}

// ---------------------------------------------------------------------------
extern "C" void kernel_launch(void* const* d_in, const int* in_sizes, int n_in,
                              void* d_out, int out_size, void* d_ws, size_t ws_size,
                              hipStream_t stream) {
  const float* hidden = (const float*)d_in[0];
  const float* sx     = (const float*)d_in[1];
  const float* aa     = (const float*)d_in[2];
  const float* bb     = (const float*)d_in[3];
  const float* pp     = (const float*)d_in[4];
  const float* tdec   = (const float*)d_in[5];
  const float* tfirst = (const float*)d_in[6];
  const float* tmk    = (const float*)d_in[7];
  const float* tmv    = (const float*)d_in[8];
  const float* tmr    = (const float*)d_in[9];
  const float* Wk     = (const float*)d_in[10];
  const float* Wv     = (const float*)d_in[11];
  const float* Wr     = (const float*)d_in[12];
  const float* Wo     = (const float*)d_in[13];
  float* out = (float*)d_out;

  const size_t TH = (size_t)T_DIM * H_DIM;
  const size_t HH = (size_t)H_DIM * H_DIM;

  char* w = (char*)d_ws;
  bf16_t* kx   = (bf16_t*)w; w += TH * 2;
  bf16_t* vx   = (bf16_t*)w; w += TH * 2;
  bf16_t* rx   = (bf16_t*)w; w += TH * 2;
  bf16_t* rwkv = (bf16_t*)w; w += TH * 2;
  bf16_t* WkT  = (bf16_t*)w; w += HH * 2;
  bf16_t* WvT  = (bf16_t*)w; w += HH * 2;
  bf16_t* WrT  = (bf16_t*)w; w += HH * 2;
  bf16_t* WoT  = (bf16_t*)w; w += HH * 2;
  float*  kf   = (float*)w;  w += TH * 4;
  float*  vf   = (float*)w;  w += TH * 4;
  float*  rf   = (float*)w;  w += TH * 4;

  mix_kernel<<<(int)(TH / 256), 256, 0, stream>>>(hidden, sx, tmk, tmv, tmr,
                                                  kx, vx, rx);
  transpose_kernel<<<dim3(H_DIM / 32, H_DIM / 32, 4), 256, 0, stream>>>(
      Wk, Wv, Wr, Wo, WkT, WvT, WrT, WoT);

  dim3 ggrid(H_DIM / BLK_N, T_DIM / BLK_M);
  gemm_kernel<<<ggrid, 256, 0, stream>>>(kx, WkT, kf, nullptr, 0);
  gemm_kernel<<<ggrid, 256, 0, stream>>>(vx, WvT, vf, nullptr, 0);
  gemm_kernel<<<ggrid, 256, 0, stream>>>(rx, WrT, rf, nullptr, 1);

  scan_kernel<<<H_DIM / 256, 256, 0, stream>>>(kf, vf, rf, aa, bb, pp,
                                               tdec, tfirst, hidden,
                                               rwkv, out + TH);

  gemm_kernel<<<ggrid, 256, 0, stream>>>(rwkv, WoT, out, hidden, 2);
}
